// VoteAttentionNeck_35502199669119
// MI455X (gfx1250) — matrix operand order repaired and checked by reading, not compiled, not run on hardware
//
#include <hip/hip_runtime.h>

// ---------------- types ----------------
typedef __attribute__((ext_vector_type(16))) __bf16        v16bf;
typedef __attribute__((ext_vector_type(8)))  float         v8f;
typedef __attribute__((ext_vector_type(4)))  unsigned int  u32x4;
typedef __attribute__((ext_vector_type(4)))  float         f32x4;

union Frag { u32x4 q[2]; v16bf v; };

// ---------------- problem constants ----------------
#define BATCH   4
#define NPT     65536
#define CH      128
#define NCLS    3
#define TOPK    2048

// ---------------- workspace layout (bytes) ----------------
#define WS_W1C   0u
#define WS_W1O   32768u
#define WS_W2C   65536u
#define WS_W2O   69632u
#define WS_BNAC  73728u          // 4 contiguous arrays of 128 f32 (AC,BC,AO,BO)
#define WS_B2C   75776u          // 16+16 f32 contiguous
#define WS_SCORE 76800u          // B*N*3 f32  = 3145728 B
#define WS_VOTES (WS_SCORE + 3145728u)  // B*N*2 f32 = 2097152 B
#define WS_TOPK  (WS_VOTES + 2097152u)  // B*K*NC int = 98304 B

// ---------------- LDS layout (bytes), row stride 272 = 256 + 16 pad ----------------
#define L_STRIDE 272
#define L_W1C  0
#define L_W1O  34816
#define L_W2C  69632
#define L_W2O  73984
#define L_BNAC 78336            // 4*512 contiguous
#define L_B2C  80384            // 32 f32
#define L_X    80512            // 128 rows * 272
#define L_H    115328           // 128 rows * 272
#define L_TOTAL 150144

// ---------------- output layout (f32 elements) ----------------
#define OUT_VOTES  0
#define OUT_SCORES 73728
#define OUT_FEATS  147456

// CDNA5 async global->LDS copy path (ASYNCcnt-tracked, bypasses VGPRs)
#if defined(__gfx1250__) && __has_builtin(__builtin_amdgcn_global_load_async_to_lds_b128)
#define HAVE_ASYNC_LDS 1
#endif

#ifdef HAVE_ASYNC_LDS
typedef __attribute__((ext_vector_type(4))) int i32v4;
#define AS_G __attribute__((address_space(1)))
#define AS_L __attribute__((address_space(3)))
__device__ __forceinline__ void asyncCopyB128(const void* g, void* l) {
    __builtin_amdgcn_global_load_async_to_lds_b128(
        (AS_G i32v4*)(void*)g, (AS_L i32v4*)l, 0, 0);
}
__device__ __forceinline__ void asyncCopyB32(const void* g, void* l) {
    __builtin_amdgcn_global_load_async_to_lds_b32(
        (AS_G int*)(void*)g, (AS_L int*)l, 0, 0);
}
#endif

__device__ __forceinline__ unsigned short f2bf(float f) {
    union { float f; unsigned u; } x; x.f = f;
    unsigned r = x.u + 0x7FFFu + ((x.u >> 16) & 1u);
    return (unsigned short)(r >> 16);
}

// =====================================================================
// Kernel 1: weight prep (transpose to bf16, pad W2, fold BN)
// =====================================================================
__global__ __launch_bounds__(256) void prep_kernel(
    const float* __restrict__ cW1, const float* __restrict__ cW2,
    const float* __restrict__ cg,  const float* __restrict__ cbb,
    const float* __restrict__ cm,  const float* __restrict__ cv,
    const float* __restrict__ cb2,
    const float* __restrict__ oW1, const float* __restrict__ oW2,
    const float* __restrict__ og,  const float* __restrict__ obb,
    const float* __restrict__ om,  const float* __restrict__ ov,
    const float* __restrict__ ob2,
    char* __restrict__ ws)
{
    int i = blockIdx.x * 256 + threadIdx.x;      // 64 blocks -> i in [0,16384)
    int n = i >> 7, k = i & 127;
    ((unsigned short*)(ws + WS_W1C))[i] = f2bf(cW1[k * CH + n]);  // W1^T[n][k]
    ((unsigned short*)(ws + WS_W1O))[i] = f2bf(oW1[k * CH + n]);
    if (i < 2048) {                              // W2 padded to 16 cols, transposed
        int n2 = i >> 7, k2 = i & 127;
        ((unsigned short*)(ws + WS_W2C))[i] = (n2 < NCLS) ? f2bf(cW2[k2 * NCLS + n2]) : (unsigned short)0;
        ((unsigned short*)(ws + WS_W2O))[i] = (n2 < 2)    ? f2bf(oW2[k2 * 2 + n2])    : (unsigned short)0;
    }
    if (i < 128) {
        float ac = cg[i] * rsqrtf(cv[i] + 1e-5f);
        ((float*)(ws + WS_BNAC))[i]         = ac;
        ((float*)(ws + WS_BNAC))[128 + i]   = cbb[i] - cm[i] * ac;
        float ao = og[i] * rsqrtf(ov[i] + 1e-5f);
        ((float*)(ws + WS_BNAC))[256 + i]   = ao;
        ((float*)(ws + WS_BNAC))[384 + i]   = obb[i] - om[i] * ao;
    }
    if (i < 16) {
        ((float*)(ws + WS_B2C))[i]      = (i < NCLS) ? cb2[i] : 0.f;
        ((float*)(ws + WS_B2C))[16 + i] = (i < 2)    ? ob2[i] : 0.f;
    }
}

// =====================================================================
// Kernel 2: fused dual-MLP, bf16 WMMA
// =====================================================================
__device__ __forceinline__ v8f run_mlp(const char* smem, const Frag a[4], char* hb,
                                       int w1, int w2, int bna, int bnb, int b2,
                                       int rowM, int hi)
{
    // ---- layer 1: x(16x128) @ W1(128x128), BN+ReLU, write h to LDS as bf16 ----
#pragma unroll 1
    for (int tn = 0; tn < 8; ++tn) {
        v8f acc = {};
#pragma unroll
        for (int kb = 0; kb < 4; ++kb) {
            Frag bq;
            int k0 = kb * 32 + hi * 16;                       // B layout: lane<16 K 0..15
            const char* wb = smem + w1 + (tn * 16 + rowM) * L_STRIDE + k0 * 2;
            bq.q[0] = *(const u32x4*)(wb);
            bq.q[1] = *(const u32x4*)(wb + 16);
            acc = __builtin_amdgcn_wmma_f32_16x16x32_bf16(false, a[kb].v, false, bq.v,
                                                          (short)0, acc, false, false);
        }
        int ch = tn * 16 + rowM;
        float aS = *(const float*)(smem + bna + ch * 4);
        float bS = *(const float*)(smem + bnb + ch * 4);
#pragma unroll
        for (int r = 0; r < 8; ++r) {
            float h = fmaxf(acc[r] * aS + bS, 0.0f);
            *(unsigned short*)(hb + (r + hi * 8) * L_STRIDE + ch * 2) = f2bf(h);
        }
    }
    // ---- layer 2: h(16x128) @ W2pad(128x16) ----
    Frag ah[4];
#pragma unroll
    for (int kb = 0; kb < 4; ++kb) {
        int k0 = kb * 32 + hi * 16;                           // A layout: K 0-15/16-31
        ah[kb].q[0] = *(const u32x4*)(hb + rowM * L_STRIDE + k0 * 2);
        ah[kb].q[1] = *(const u32x4*)(hb + rowM * L_STRIDE + k0 * 2 + 16);
    }
    v8f s = {};
#pragma unroll
    for (int kb = 0; kb < 4; ++kb) {
        Frag bq;
        int k0 = kb * 32 + hi * 16;
        const char* wb = smem + w2 + rowM * L_STRIDE + k0 * 2;
        bq.q[0] = *(const u32x4*)wb;
        bq.q[1] = *(const u32x4*)(wb + 16);
        s = __builtin_amdgcn_wmma_f32_16x16x32_bf16(false, ah[kb].v, false, bq.v,
                                                    (short)0, s, false, false);
    }
    float b2v = *(const float*)(smem + b2 + rowM * 4);
#pragma unroll
    for (int r = 0; r < 8; ++r) s[r] += b2v;
    return s;
}

__global__ __launch_bounds__(256) void mlp_kernel(
    const float* __restrict__ feats, const int* __restrict__ indices,
    const int* __restrict__ strideP, char* __restrict__ ws)
{
    extern __shared__ char smem[];
    const int tid = threadIdx.x;
    const int blockBase = blockIdx.x * 128;

    // stage weights (bf16) into padded LDS: 4608 chunks of 16B
    for (int i = tid; i < 4608; i += 256) {
        const char* src; char* dst; int local;
        if (i < 2048)       { local = i;        src = ws + WS_W1C; dst = smem + L_W1C; }
        else if (i < 4096)  { local = i - 2048; src = ws + WS_W1O; dst = smem + L_W1O; }
        else if (i < 4352)  { local = i - 4096; src = ws + WS_W2C; dst = smem + L_W2C; }
        else                { local = i - 4352; src = ws + WS_W2O; dst = smem + L_W2O; }
        int rown = local >> 4, kc = local & 15;
#ifdef HAVE_ASYNC_LDS
        asyncCopyB128(src + rown * 256 + kc * 16, dst + rown * L_STRIDE + kc * 16);
#else
        *(u32x4*)(dst + rown * L_STRIDE + kc * 16) = *(const u32x4*)(src + rown * 256 + kc * 16);
#endif
    }
    // BN tables: 512 f32 contiguous
    for (int i = tid; i < 512; i += 256) {
#ifdef HAVE_ASYNC_LDS
        asyncCopyB32(ws + WS_BNAC + i * 4, smem + L_BNAC + i * 4);
#else
        *(float*)(smem + L_BNAC + i * 4) = *(const float*)(ws + WS_BNAC + i * 4);
#endif
    }
    if (tid < 32)
        *(float*)(smem + L_B2C + tid * 4) = *(const float*)(ws + WS_B2C + tid * 4);

    // stage 128x128 f32 feature tile -> bf16 LDS (conversion needs VGPRs anyway)
#pragma unroll 4
    for (int it = 0; it < 16; ++it) {
        int i = tid + it * 256;                 // [0,4096)
        int row = i >> 5, c4 = i & 31;
        f32x4 v = *(const f32x4*)(feats + (size_t)(blockBase + row) * CH + c4 * 4);
        unsigned long long pk =
            (unsigned long long)f2bf(v[0])
          | ((unsigned long long)f2bf(v[1]) << 16)
          | ((unsigned long long)f2bf(v[2]) << 32)
          | ((unsigned long long)f2bf(v[3]) << 48);
        *(unsigned long long*)(smem + L_X + row * L_STRIDE + c4 * 8) = pk;
    }

#ifdef HAVE_ASYNC_LDS
#if __has_builtin(__builtin_amdgcn_s_wait_asynccnt)
    __builtin_amdgcn_s_wait_asynccnt(0);
#else
    asm volatile("s_wait_asynccnt 0" ::: "memory");
#endif
#endif
    __syncthreads();

    const int lane = tid & 31, wv = tid >> 5;
    const int rowM = lane & 15;
    const int hi   = (lane >> 4) & 1;
    const char* xb = smem + L_X + (wv * 16) * L_STRIDE;
    char*       hb = smem + L_H + (wv * 16) * L_STRIDE;

    Frag a[4];
#pragma unroll
    for (int kb = 0; kb < 4; ++kb) {
        int k0 = kb * 32 + hi * 16;
        a[kb].q[0] = *(const u32x4*)(xb + rowM * L_STRIDE + k0 * 2);
        a[kb].q[1] = *(const u32x4*)(xb + rowM * L_STRIDE + k0 * 2 + 16);
    }
    float offScale = 16.0f / (float)strideP[0];

    // classification path
    v8f sc = run_mlp(smem, a, hb, L_W1C, L_W2C, L_BNAC, L_BNAC + 512, L_B2C, rowM, hi);
    if (rowM < NCLS) {
#pragma unroll
        for (int r = 0; r < 8; ++r) {
            int prow = blockBase + wv * 16 + r + hi * 8;
            *(float*)(ws + WS_SCORE + ((size_t)prow * 3 + rowM) * 4) = sc[r];
        }
    }
    // offset path (reuses h LDS area; wave-local in-order DS keeps this safe)
    v8f so = run_mlp(smem, a, hb, L_W1O, L_W2O, L_BNAC + 1024, L_BNAC + 1536, L_B2C + 64, rowM, hi);
    if (rowM < 2) {
#pragma unroll
        for (int r = 0; r < 8; ++r) {
            int prow = blockBase + wv * 16 + r + hi * 8;
            int iv = indices[(size_t)prow * 3 + 1 + rowM];
            float lim = ceilf(so[r] * offScale);
            lim = fminf(3.0f, fmaxf(-3.0f, lim));
            *(float*)(ws + WS_VOTES + ((size_t)prow * 2 + rowM) * 4) = (float)iv + lim;
        }
    }
}

// =====================================================================
// Kernel 3: per-(batch,class) top-K via radix select + bitonic sort
// =====================================================================
__device__ __forceinline__ unsigned keyOf(float f) {
    union { float f; unsigned u; } x; x.f = f;
    return (x.u & 0x80000000u) ? ~x.u : (x.u | 0x80000000u);
}

__global__ __launch_bounds__(1024) void topk_kernel(const float* __restrict__ score,
                                                    int* __restrict__ topk)
{
    __shared__ unsigned hist[256];
    __shared__ unsigned long long vals[TOPK];
    __shared__ unsigned cG, cE, sPrefix, sKRem;

    const int b = blockIdx.x / NCLS, c = blockIdx.x % NCLS;
    const int tid = threadIdx.x;
    const float* sc = score + (size_t)b * NPT * 3 + c;

    unsigned prefix = 0;
    int kRem = TOPK;
    for (int pass = 0; pass < 4; ++pass) {
        int shift = 24 - 8 * pass;
        for (int i = tid; i < 256; i += 1024) hist[i] = 0;
        __syncthreads();
        for (int n = tid; n < NPT; n += 1024) {
            unsigned key = keyOf(sc[(size_t)n * 3]);
            if (pass == 0 || (key >> (shift + 8)) == prefix)
                atomicAdd(&hist[(key >> shift) & 255u], 1u);
        }
        __syncthreads();
        if (tid == 0) {
            int cum = 0, t = 255;
            for (; t > 0; --t) { cum += (int)hist[t]; if (cum >= kRem) break; }
            if (cum < kRem) cum += (int)hist[0];                 // t == 0
            int above = cum - (int)hist[t];
            sPrefix = (prefix << 8) | (unsigned)t;
            sKRem   = (unsigned)(kRem - above);
        }
        __syncthreads();
        prefix = sPrefix; kRem = (int)sKRem;
        __syncthreads();
    }
    const unsigned T = prefix;
    const int countGreater = TOPK - kRem;
    if (tid == 0) { cG = 0; cE = 0; }
    __syncthreads();
    for (int n = tid; n < NPT; n += 1024) {
        unsigned key = keyOf(sc[(size_t)n * 3]);
        if (key > T) {
            unsigned p = atomicAdd(&cG, 1u);
            vals[p] = ~(((unsigned long long)key << 32) | (unsigned long long)(0xFFFFFFFFu - (unsigned)n));
        } else if (key == T) {
            unsigned p = atomicAdd(&cE, 1u);
            if ((int)p < kRem)
                vals[countGreater + p] = ~(((unsigned long long)key << 32) | (unsigned long long)(0xFFFFFFFFu - (unsigned)n));
        }
    }
    __syncthreads();
    // bitonic ascending sort of complemented keys -> descending scores, asc index tiebreak
    for (int k2 = 2; k2 <= TOPK; k2 <<= 1)
        for (int j = k2 >> 1; j > 0; j >>= 1) {
            for (int i = tid; i < TOPK; i += 1024) {
                int ixj = i ^ j;
                if (ixj > i) {
                    bool up = ((i & k2) == 0);
                    unsigned long long av = vals[i], bv = vals[ixj];
                    if ((av > bv) == up) { vals[i] = bv; vals[ixj] = av; }
                }
            }
            __syncthreads();
        }
    for (int i = tid; i < TOPK; i += 1024) {
        unsigned long long v = ~vals[i];
        unsigned n = 0xFFFFFFFFu - (unsigned)(v & 0xFFFFFFFFull);
        topk[((size_t)b * TOPK + i) * NCLS + c] = (int)n;        // [b][k][c]
    }
}

// =====================================================================
// Kernel 4: gather tk_votes / tk_scores / tk_features
// =====================================================================
__global__ __launch_bounds__(256) void gather_kernel(
    const float* __restrict__ feats, const int* __restrict__ topk,
    const float* __restrict__ scoreWs, const float* __restrict__ votesWs,
    float* __restrict__ out)
{
    int row  = blockIdx.x * 8 + (threadIdx.x >> 5);
    int lane = threadIdx.x & 31;
    if (row >= BATCH * TOPK * NCLS) return;
    int b = row / (TOPK * NCLS);
    int n = topk[row];
    size_t p = (size_t)b * NPT + (size_t)n;

    f32x4 v = *(const f32x4*)(feats + p * CH + lane * 4);
    *(f32x4*)(out + OUT_FEATS + (size_t)row * CH + lane * 4) = v;

    if (lane == 0) {
        out[OUT_VOTES  + row * 3 + 0] = (float)b;
        out[OUT_VOTES  + row * 3 + 1] = votesWs[p * 2 + 0];
        out[OUT_VOTES  + row * 3 + 2] = votesWs[p * 2 + 1];
        out[OUT_SCORES + row * 3 + 0] = scoreWs[p * 3 + 0];
        out[OUT_SCORES + row * 3 + 1] = scoreWs[p * 3 + 1];
        out[OUT_SCORES + row * 3 + 2] = scoreWs[p * 3 + 2];
    }
}

// =====================================================================
extern "C" void kernel_launch(void* const* d_in, const int* in_sizes, int n_in,
                              void* d_out, int out_size, void* d_ws, size_t ws_size,
                              hipStream_t stream)
{
    (void)in_sizes; (void)n_in; (void)out_size; (void)ws_size;
    const float* features = (const float*)d_in[0];
    const int*   indices  = (const int*)d_in[1];
    const int*   strideP  = (const int*)d_in[2];
    const float* cW1 = (const float*)d_in[3];
    const float* cg  = (const float*)d_in[4];
    const float* cbb = (const float*)d_in[5];
    const float* cm  = (const float*)d_in[6];
    const float* cv  = (const float*)d_in[7];
    const float* cW2 = (const float*)d_in[8];
    const float* cb2 = (const float*)d_in[9];
    const float* oW1 = (const float*)d_in[10];
    const float* og  = (const float*)d_in[11];
    const float* obb = (const float*)d_in[12];
    const float* om  = (const float*)d_in[13];
    const float* ov  = (const float*)d_in[14];
    const float* oW2 = (const float*)d_in[15];
    const float* ob2 = (const float*)d_in[16];
    char*  ws  = (char*)d_ws;
    float* out = (float*)d_out;

    prep_kernel<<<64, 256, 0, stream>>>(cW1, cW2, cg, cbb, cm, cv, cb2,
                                        oW1, oW2, og, obb, om, ov, ob2, ws);
    mlp_kernel<<<(BATCH * NPT) / 128, 256, L_TOTAL, stream>>>(features, indices, strideP, ws);
    topk_kernel<<<BATCH * NCLS, 1024, 0, stream>>>((const float*)(ws + WS_SCORE),
                                                   (int*)(ws + WS_TOPK));
    gather_kernel<<<(BATCH * TOPK * NCLS) / 8, 256, 0, stream>>>(
        features, (const int*)(ws + WS_TOPK),
        (const float*)(ws + WS_SCORE), (const float*)(ws + WS_VOTES), out);
}
